// MrLoraLayer_62697932587275
// MI455X (gfx1250) — matrix operand order repaired and checked by reading, not compiled, not run on hardware
//
#include <hip/hip_runtime.h>
#include <hip/hip_bf16.h>
#include <stdint.h>

// ---------------------------------------------------------------------------
// y = x @ (W + 0.5*(a0*B8A8 + a1*B16A16 + a2*B32A32))^T
// M=8192, N=K=4096.  bf16 WMMA GEMM, f32 accum.
// Block tile 128x256, 8 waves, wave tile 64x64 (16 WMMA per k-stage per wave).
// ---------------------------------------------------------------------------

typedef __attribute__((ext_vector_type(16))) __bf16 v16bf;
typedef __attribute__((ext_vector_type(8)))  float  v8f;

#define DDIM 4096
#define MDIM 8192
#define BM   128
#define BN   256
#define BK   32
#define LDST 40              // BK + 8 pad (elements); 80B row stride, conflict-free
#define KT   (DDIM / BK)     // 128 k-stages

union Frag { uint4 q[2]; v16bf v; };

__device__ __forceinline__ unsigned short f2bf(float f) {
  unsigned u = __builtin_bit_cast(unsigned, f);
  u += 0x7FFFu + ((u >> 16) & 1u);             // round-to-nearest-even
  return (unsigned short)(u >> 16);
}

// ---- kernel 1: x fp32 -> bf16 (8 elements / thread) -----------------------
__global__ void k_cvt_x(const float* __restrict__ x, unsigned short* __restrict__ xb) {
  int t = blockIdx.x * blockDim.x + threadIdx.x;
  const float4* xv = (const float4*)x;
  float4 a = xv[2 * t], b = xv[2 * t + 1];
  uint4 o;
  o.x = (unsigned)f2bf(a.x) | ((unsigned)f2bf(a.y) << 16);
  o.y = (unsigned)f2bf(a.z) | ((unsigned)f2bf(a.w) << 16);
  o.z = (unsigned)f2bf(b.x) | ((unsigned)f2bf(b.y) << 16);
  o.w = (unsigned)f2bf(b.z) | ((unsigned)f2bf(b.w) << 16);
  ((uint4*)xb)[t] = o;
}

// ---- kernel 2: W_eff = W + 0.5*sum(alpha_i * B_i A_i) -> bf16 -------------
__global__ void k_weff(const float* __restrict__ W,
                       const float* __restrict__ A8,  const float* __restrict__ B8,
                       const float* __restrict__ A16, const float* __restrict__ B16,
                       const float* __restrict__ A32, const float* __restrict__ B32,
                       const float* __restrict__ alphas,
                       unsigned short* __restrict__ wb) {
  const int o = blockIdx.y;                         // output row (uniform per block)
  const int k = blockIdx.x * 1024 + threadIdx.x * 4;
  const float s0 = 0.5f * alphas[0];
  const float s1 = 0.5f * alphas[1];
  const float s2 = 0.5f * alphas[2];
  float4 acc = *(const float4*)&W[o * DDIM + k];
#pragma unroll
  for (int r = 0; r < 8; ++r) {
    float c = s0 * B8[o * 8 + r];
    float4 av = *(const float4*)&A8[r * DDIM + k];
    acc.x += c * av.x; acc.y += c * av.y; acc.z += c * av.z; acc.w += c * av.w;
  }
#pragma unroll
  for (int r = 0; r < 16; ++r) {
    float c = s1 * B16[o * 16 + r];
    float4 av = *(const float4*)&A16[r * DDIM + k];
    acc.x += c * av.x; acc.y += c * av.y; acc.z += c * av.z; acc.w += c * av.w;
  }
#pragma unroll
  for (int r = 0; r < 32; ++r) {
    float c = s2 * B32[o * 32 + r];
    float4 av = *(const float4*)&A32[r * DDIM + k];
    acc.x += c * av.x; acc.y += c * av.y; acc.z += c * av.z; acc.w += c * av.w;
  }
  uint2 p;
  p.x = (unsigned)f2bf(acc.x) | ((unsigned)f2bf(acc.y) << 16);
  p.y = (unsigned)f2bf(acc.z) | ((unsigned)f2bf(acc.w) << 16);
  *(uint2*)&wb[o * DDIM + k] = p;
}

// ---- kernel 3: out[M,N] = xb[M,K] * wb[N,K]^T  (bf16 WMMA, f32 accum) -----
__global__ void __launch_bounds__(256)
k_gemm(const unsigned short* __restrict__ xb,
       const unsigned short* __restrict__ wb,
       float* __restrict__ out) {
  extern __shared__ unsigned short smem[];   // double-buffered A/B tiles

  const int tid   = threadIdx.x;
  const int lane  = tid & 31;
  const int wid   = tid >> 5;            // 8 waves
  const int waveM = wid >> 2;            // 0..1  -> 64-row strip
  const int waveN = wid & 3;             // 0..3  -> 64-col strip
  const int lr    = lane & 15;
  const int half  = lane >> 4;
  const int m0 = blockIdx.y * BM;
  const int n0 = blockIdx.x * BN;

  // LDS layout (element offsets): A buf0 | A buf1 | B buf0 | B buf1
  const int AS[2] = {0, BM * LDST};                            // 0, 5120
  const int BS[2] = {2 * BM * LDST, 2 * BM * LDST + BN * LDST};// 10240, 20480
  // total = (2*128 + 2*256) * 40 * 2B = 61440 bytes

  // Issue one k-stage of async global->LDS copies (6 x b128 per thread).
  auto issue = [&](int t, int b) {
    const int kB = t * BK;
#pragma unroll
    for (int i = 0; i < 2; ++i) {                      // A tile: 128 x 32 bf16
      int chunk = tid + i * 256;                       // 512 chunks of 16B
      int row = chunk >> 2, cir = chunk & 3;
      unsigned goff = (unsigned)(((m0 + row) * DDIM + kB + cir * 8) * 2);
      unsigned loff = (unsigned)((AS[b] + row * LDST + cir * 8) * 2);
      asm volatile("global_load_async_to_lds_b128 %0, %1, %2"
                   :: "v"(loff), "v"(goff), "s"(xb) : "memory");
    }
#pragma unroll
    for (int i = 0; i < 4; ++i) {                      // B tile: 256 x 32 bf16
      int chunk = tid + i * 256;                       // 1024 chunks of 16B
      int row = chunk >> 2, cir = chunk & 3;
      unsigned goff = (unsigned)(((n0 + row) * DDIM + kB + cir * 8) * 2);
      unsigned loff = (unsigned)((BS[b] + row * LDST + cir * 8) * 2);
      asm volatile("global_load_async_to_lds_b128 %0, %1, %2"
                   :: "v"(loff), "v"(goff), "s"(wb) : "memory");
    }
  };

  v8f acc[4][4] = {};

  issue(0, 0);
  for (int t = 0; t < KT; ++t) {
    const int cur = t & 1;
    if (t + 1 < KT) {
      issue(t + 1, cur ^ 1);                 // prefetch next stage (6 loads)
      asm volatile("s_wait_asynccnt 6" ::: "memory");  // current stage landed
    } else {
      asm volatile("s_wait_asynccnt 0" ::: "memory");
    }
    __syncthreads();

    const int aB = AS[cur], bB = BS[cur];
    // A fragments: lane<16 -> K 0..7 & 16..23 ; lane>=16 -> K 8..15 & 24..31
    Frag af[4];
#pragma unroll
    for (int i = 0; i < 4; ++i) {
      int eb = aB + (waveM * 64 + i * 16 + lr) * LDST + half * 8;
      af[i].q[0] = *(const uint4*)&smem[eb];
      af[i].q[1] = *(const uint4*)&smem[eb + 16];
    }
    // B fragments: lane<16 -> K 0..7 & 16..23 of col lr ; lane>=16 -> K 8..15 & 24..31
    Frag bfr[4];
#pragma unroll
    for (int j = 0; j < 4; ++j) {
      int eb = bB + (waveN * 64 + j * 16 + lr) * LDST + half * 8;
      bfr[j].q[0] = *(const uint4*)&smem[eb];
      bfr[j].q[1] = *(const uint4*)&smem[eb + 16];
    }
#pragma unroll
    for (int i = 0; i < 4; ++i)
#pragma unroll
      for (int j = 0; j < 4; ++j)
        acc[i][j] = __builtin_amdgcn_wmma_f32_16x16x32_bf16(
            false, af[i].v, false, bfr[j].v, (short)0, acc[i][j], false, false);

    __syncthreads();
  }

  // C/D layout: vgpr r -> row r + 8*half, lane%16 -> col
#pragma unroll
  for (int i = 0; i < 4; ++i)
#pragma unroll
    for (int j = 0; j < 4; ++j) {
      int r0 = m0 + waveM * 64 + i * 16 + half * 8;
      int c  = n0 + waveN * 64 + j * 16 + lr;
#pragma unroll
      for (int r = 0; r < 8; ++r)
        out[(r0 + r) * DDIM + c] = acc[i][j][r];
    }
}

// ---------------------------------------------------------------------------
extern "C" void kernel_launch(void* const* d_in, const int* in_sizes, int n_in,
                              void* d_out, int out_size, void* d_ws, size_t ws_size,
                              hipStream_t stream) {
  (void)in_sizes; (void)n_in; (void)out_size; (void)ws_size;
  const float* x      = (const float*)d_in[0];
  const float* W      = (const float*)d_in[1];
  const float* A8     = (const float*)d_in[2];
  const float* B8     = (const float*)d_in[3];
  const float* A16    = (const float*)d_in[4];
  const float* B16    = (const float*)d_in[5];
  const float* A32    = (const float*)d_in[6];
  const float* B32    = (const float*)d_in[7];
  const float* alphas = (const float*)d_in[8];
  float* out = (float*)d_out;

  unsigned short* xb = (unsigned short*)d_ws;                  // 8192*4096 bf16 (64 MB)
  unsigned short* wb = xb + (size_t)MDIM * DDIM;               // 4096*4096 bf16 (32 MB)

  k_cvt_x<<<(MDIM * DDIM) / (256 * 8), 256, 0, stream>>>(x, xb);
  k_weff<<<dim3(DDIM / 1024, DDIM), 256, 0, stream>>>(W, A8, B8, A16, B16, A32, B32,
                                                      alphas, wb);
  k_gemm<<<dim3(DDIM / BN, MDIM / BM), 256,
           (2 * BM + 2 * BN) * LDST * 2, stream>>>(xb, wb, out);
}
